// ODE_Decoder_83691732730093
// MI455X (gfx1250) — matrix operand order repaired and checked by reading, not compiled, not run on hardware
//
#include <hip/hip_runtime.h>
#include <hip/hip_bf16.h>
#include <stdint.h>

typedef unsigned short u16;
typedef __attribute__((ext_vector_type(16))) __bf16 v16bf;
typedef __attribute__((ext_vector_type(8)))  float  v8f;

constexpr int Bc  = 2;
constexpr int Lc  = 1024;
constexpr int Dc  = 512;
constexpr int Hc  = 8;
constexpr int DKc = 64;
constexpr int DFFc = 2048;
constexpr int BL = Bc * Lc;   // 2048 rows of activations
constexpr int BH = Bc * Hc;   // 16 attention batches

// ---- CDNA5 async global->LDS copy (no VGPR staging, tracked by ASYNCcnt) ----
// Builtin signature (from clang diagnostic): (int4 __device__*, int4 __shared__*,
// imm offset, imm cpol) with GCC-vector int4 pointees.
typedef int gvi4 __attribute__((vector_size(16)));

#if defined(__gfx1250__) && __has_builtin(__builtin_amdgcn_global_load_async_to_lds_b128)
#define USE_ASYNC_LDS 1
#define GL2LDS_B128(g, l)                                                     \
  __builtin_amdgcn_global_load_async_to_lds_b128(                             \
      (__attribute__((address_space(1))) gvi4*)(g),                           \
      (__attribute__((address_space(3))) gvi4*)(l), 0, 0)
#else
#define USE_ASYNC_LDS 0
#endif

__device__ __forceinline__ void wait_async_lds() {
#if USE_ASYNC_LDS
#if __has_builtin(__builtin_amdgcn_s_wait_asynccnt)
  __builtin_amdgcn_s_wait_asynccnt(0);
#else
  asm volatile("s_wait_asynccnt 0x0" ::: "memory");
#endif
#endif
}

// ---------------------------------------------------------------- helpers

__device__ __forceinline__ u16 f2bf(float f) {
  unsigned u = __float_as_uint(f);
  unsigned r = (u + 0x7FFFu + ((u >> 16) & 1u)) >> 16;   // round-nearest-even
  return (u16)r;
}

// ---------------------------------------------------------------- GEMM
// C[M,N] = scale * A[M,K](bf16) x B[K,N](bf16) (+bias) (+resid) (relu?)
// TRANSB==1: B given as Bt[N,K] row-major (e.g. Q*K^T with per-head K rows)
// OUTMODE: 0 f32 plain  1 bf16 plain  2 bf16 head-split  3 bf16 head-merge
//
// Block tile 256x64 (8 waves, 4Mx2N), wave tile 64x32 = 4x2 WMMA accs.
// Two-stage pipeline: chunk k+1 streams into LDS buf^1 via
// global_load_async_to_lds_b128 (A tile; zero VGPR staging) while chunk k is
// computed from buf; B tile stages through 1 uint4 (scatter-transpose needs
// registers). s_wait_asynccnt 0 + barrier closes each iteration.

#define BM 256
#define BN 64
#define BK 32
#define LDP (BK + 8)   // row pitch in u16: 80B, keeps 16B alignment, skews banks

union Frag { uint4 q[2]; v16bf v; };

template <int TRANSB, int OUTMODE>
__global__ __launch_bounds__(256, 1)
void gemm_bf16_kernel(const u16* __restrict__ A, int lda, long long sA,
                      const u16* __restrict__ Bm, int ldb, long long sB,
                      const float* __restrict__ bias,
                      const float* __restrict__ resid,
                      float* __restrict__ outF, u16* __restrict__ outB, long long sC,
                      int M, int N, int K, float scale, int relu)
{
  __shared__ u16 lA [2][BM][LDP];   // 2 x 256 x 32 bf16 (double-buffered)
  __shared__ u16 lBt[2][BN][LDP];   // B stored N-major (rows = output columns)

  const int z = blockIdx.z;
  A  += (long long)z * sA;
  Bm += (long long)z * sB;

  const int m0   = blockIdx.y * BM;
  const int n0   = blockIdx.x * BN;
  const int tid  = threadIdx.x;     // 256 threads = 8 waves (wave32)
  const int wave = tid >> 5;
  const int lane = tid & 31;

  // ---- staging -----------------------------------------------------------
  // A: one thread per row, 32 bf16 = 4x b128 per thread.
  const u16* aRow = A + (long long)(m0 + tid) * lda;
  // B(trans): 64 rows x 32, 4 threads/row, 1x b128 each.
  const int btRow = tid >> 2, btQd = tid & 3;
  // B(plain): 32 rows x 64, 8 threads/row, 8 bf16 each, scatter-transpose.
  const int bKrow = tid >> 3, bSeg = tid & 7;

#if !USE_ASYNC_LDS
  uint4 aReg[4];
#endif
  uint4 bReg;

  auto loadA = [&](int kk, int buf) {
#if USE_ASYNC_LDS
    #pragma unroll
    for (int e = 0; e < 4; ++e)
      GL2LDS_B128(aRow + kk + e * 8, &lA[buf][tid][e * 8]);   // global -> LDS direct
#else
    (void)buf;
    #pragma unroll
    for (int e = 0; e < 4; ++e) aReg[e] = *(const uint4*)(aRow + kk + e * 8);
#endif
  };
  auto storeA = [&](int buf) {
#if !USE_ASYNC_LDS
    #pragma unroll
    for (int e = 0; e < 4; ++e) *(uint4*)&lA[buf][tid][e * 8] = aReg[e];
#else
    (void)buf;
#endif
  };
  auto loadB = [&](int kk) {
    if (TRANSB) {
      bReg = *(const uint4*)(Bm + (long long)(n0 + btRow) * ldb + kk + btQd * 8);
    } else {
      bReg = *(const uint4*)(Bm + (long long)(kk + bKrow) * ldb + n0 + bSeg * 8);
    }
  };
  auto storeB = [&](int buf) {
    if (TRANSB) {
      *(uint4*)&lBt[buf][btRow][btQd * 8] = bReg;
    } else {
      u16 t[8];
      *(uint4*)t = bReg;
      #pragma unroll
      for (int e = 0; e < 8; ++e) lBt[buf][bSeg * 8 + e][bKrow] = t[e];
    }
  };

  // ---- pipelined main loop ----------------------------------------------
  v8f acc[4][2] = {};
  const int wm = (wave >> 1) * 64, wn = (wave & 1) * 32;
  const int ml = lane & 15, kh = lane >> 4;

  loadA(0, 0);
  storeA(0);
  loadB(0);
  storeB(0);
  wait_async_lds();
  __syncthreads();

  int cur = 0;
  for (int k0 = 0; k0 < K; k0 += BK) {
    const bool more = (k0 + BK) < K;
    if (more) {
      loadA(k0 + BK, cur ^ 1);    // async path writes LDS buf^1 directly
      loadB(k0 + BK);
    }
    if (k0 + 2 * BK < K) __builtin_prefetch(aRow + k0 + 2 * BK, 0, 1); // global_prefetch_b8

    // fragment gather per CDNA5 16-bit A layout:
    //   lanes 0-15: K 0..7 (v0..3) and 16..23 (v4..7); lanes 16-31: +8
    Frag bfr[2];
    #pragma unroll
    for (int j = 0; j < 2; ++j) {
      const u16* pb = &lBt[cur][wn + j * 16 + ml][kh * 8];
      bfr[j].q[0] = *(const uint4*)pb;          // ds_load_b128
      bfr[j].q[1] = *(const uint4*)(pb + 16);
    }
    #pragma unroll
    for (int i = 0; i < 4; ++i) {
      Frag af;
      const u16* pa = &lA[cur][wm + i * 16 + ml][kh * 8];
      af.q[0] = *(const uint4*)pa;
      af.q[1] = *(const uint4*)(pa + 16);
      #pragma unroll
      for (int j = 0; j < 2; ++j)
        acc[i][j] = __builtin_amdgcn_wmma_f32_16x16x32_bf16(
            false, af.v, false, bfr[j].v, (short)0, acc[i][j], false, false);
    }

    if (more) {
      storeA(cur ^ 1);            // no-op on async path
      storeB(cur ^ 1);
    }
    wait_async_lds();             // ASYNCcnt==0 before handing buf^1 to others
    __syncthreads();
    cur ^= 1;
  }

  // ---- fused epilogue: C/D layout = VGPR r -> M=r (lanes 0-15), M=r+8 (16-31)
  const int nl = lane & 15, mh = lane >> 4;
  #pragma unroll
  for (int i = 0; i < 4; ++i)
    #pragma unroll
    for (int j = 0; j < 2; ++j)
      #pragma unroll
      for (int r = 0; r < 8; ++r) {
        int gm = m0 + wm + i * 16 + mh * 8 + r;
        int gn = n0 + wn + j * 16 + nl;
        float v = acc[i][j][r] * scale;
        if (bias) v += bias[gn];
        if (relu) v = fmaxf(v, 0.0f);
        if (OUTMODE == 0) {
          if (resid) v += resid[(long long)gm * N + gn];
          outF[(long long)z * sC + (long long)gm * N + gn] = v;
        } else if (OUTMODE == 1) {
          outB[(long long)z * sC + (long long)gm * N + gn] = f2bf(v);
        } else if (OUTMODE == 2) { // [B*L, D] -> [B,H,L,DK] (projection output)
          int b = gm / Lc, l = gm % Lc;
          int h = gn / DKc, dk = gn % DKc;
          outB[(((long long)b * Hc + h) * Lc + l) * DKc + dk] = f2bf(v);
        } else {                   // batched [L,DK] -> [B,L,D] (head merge)
          int b = z / Hc, h = z % Hc;
          outB[((long long)(b * Lc + gm)) * Dc + h * DKc + gn] = f2bf(v);
        }
      }
}

// ---------------------------------------------------------------- layernorm
// std with ddof=1 and eps added to std (matches reference)

template <bool BF16OUT>
__global__ __launch_bounds__(256)
void ln_kernel(const float* __restrict__ x, const float* __restrict__ g,
               const float* __restrict__ b, u16* __restrict__ outB,
               float* __restrict__ outF)
{
  __shared__ float red[256];
  const int row = blockIdx.x;
  const float* xr = x + (long long)row * Dc;
  const int t = threadIdx.x;
  float v0 = xr[t], v1 = xr[t + 256];
  red[t] = v0 + v1;
  __syncthreads();
  for (int o = 128; o > 0; o >>= 1) { if (t < o) red[t] += red[t + o]; __syncthreads(); }
  float mean = red[0] * (1.0f / Dc);
  __syncthreads();
  float d0 = v0 - mean, d1 = v1 - mean;
  red[t] = d0 * d0 + d1 * d1;
  __syncthreads();
  for (int o = 128; o > 0; o >>= 1) { if (t < o) red[t] += red[t + o]; __syncthreads(); }
  float inv = 1.0f / (sqrtf(red[0] / (float)(Dc - 1)) + 1e-6f);
  float o0 = g[t]       * d0 * inv + b[t];
  float o1 = g[t + 256] * d1 * inv + b[t + 256];
  if (BF16OUT) {
    outB[(long long)row * Dc + t]       = f2bf(o0);
    outB[(long long)row * Dc + t + 256] = f2bf(o1);
  } else {
    outF[(long long)row * Dc + t]       = o0;
    outF[(long long)row * Dc + t + 256] = o1;
  }
}

// ---------------------------------------------------------------- softmax
// masks are all-true in this problem; row length = Lc

__global__ __launch_bounds__(256)
void softmax_kernel(const float* __restrict__ S, u16* __restrict__ att)
{
  __shared__ float red[256];
  const long long row = blockIdx.x;
  const float* sr = S + row * Lc;
  const int t = threadIdx.x;
  float v[4];
  #pragma unroll
  for (int e = 0; e < 4; ++e) v[e] = sr[t + e * 256];
  float mx = fmaxf(fmaxf(v[0], v[1]), fmaxf(v[2], v[3]));
  red[t] = mx;
  __syncthreads();
  for (int o = 128; o > 0; o >>= 1) { if (t < o) red[t] = fmaxf(red[t], red[t + o]); __syncthreads(); }
  mx = red[0];
  __syncthreads();
  float s = 0.0f;
  #pragma unroll
  for (int e = 0; e < 4; ++e) { v[e] = __expf(v[e] - mx); s += v[e]; }
  red[t] = s;
  __syncthreads();
  for (int o = 128; o > 0; o >>= 1) { if (t < o) red[t] += red[t + o]; __syncthreads(); }
  float rinv = 1.0f / red[0];
  #pragma unroll
  for (int e = 0; e < 4; ++e) att[row * Lc + t + e * 256] = f2bf(v[e] * rinv);
}

// ---------------------------------------------------------------- elementwise

__global__ void f32_to_bf16_kernel(const float* __restrict__ in, u16* __restrict__ out,
                                   long long n)
{
  long long i = (long long)blockIdx.x * blockDim.x + threadIdx.x;
  long long st = (long long)gridDim.x * blockDim.x;
  for (; i < n; i += st) out[i] = f2bf(in[i]);
}

__global__ void axpy_kernel(const float* __restrict__ a, const float* __restrict__ k,
                            float c, float* __restrict__ out, long long n)
{
  long long i = (long long)blockIdx.x * blockDim.x + threadIdx.x;
  long long st = (long long)gridDim.x * blockDim.x;
  for (; i < n; i += st) out[i] = a[i] + c * k[i];
}

__global__ void rk4_combine_kernel(float* __restrict__ x,
                                   const float* __restrict__ k1, const float* __restrict__ k2,
                                   const float* __restrict__ k3, const float* __restrict__ k4,
                                   float h6, long long n)
{
  long long i = (long long)blockIdx.x * blockDim.x + threadIdx.x;
  long long st = (long long)gridDim.x * blockDim.x;
  for (; i < n; i += st) x[i] += h6 * (k1[i] + 2.0f * (k2[i] + k3[i]) + k4[i]);
}

// ---------------------------------------------------------------- host side

struct WS {
  float *xbuf, *xstage, *k1, *k2, *k3, *k4, *x1, *x2, *scores;
  u16 *n_bf, *q_bf, *k_bf, *v_bf, *attnout_bf, *mem_bf, *kmem_bf, *vmem_bf, *att_bf, *ffn_bf;
  u16 *w_attn, *w_ff1, *w_ff2;
  const float *attn_b, *ff_b1, *ff_b2, *ln_g, *ln_b;
};

static void gemm(hipStream_t st, const u16* A, int lda, long long sA,
                 const u16* Bm, int ldb, long long sB, int transB,
                 const float* bias, const float* resid,
                 float* outF, u16* outB, long long sC,
                 int M, int N, int K, int batch, float scale, int relu, int outMode)
{
  dim3 g(N / BN, M / BM, batch), blk(256);
  if (transB) {
    gemm_bf16_kernel<1, 0><<<g, blk, 0, st>>>(A, lda, sA, Bm, ldb, sB, bias, resid,
                                              outF, outB, sC, M, N, K, scale, relu);
  } else if (outMode == 0) {
    gemm_bf16_kernel<0, 0><<<g, blk, 0, st>>>(A, lda, sA, Bm, ldb, sB, bias, resid,
                                              outF, outB, sC, M, N, K, scale, relu);
  } else if (outMode == 1) {
    gemm_bf16_kernel<0, 1><<<g, blk, 0, st>>>(A, lda, sA, Bm, ldb, sB, bias, resid,
                                              outF, outB, sC, M, N, K, scale, relu);
  } else if (outMode == 2) {
    gemm_bf16_kernel<0, 2><<<g, blk, 0, st>>>(A, lda, sA, Bm, ldb, sB, bias, resid,
                                              outF, outB, sC, M, N, K, scale, relu);
  } else {
    gemm_bf16_kernel<0, 3><<<g, blk, 0, st>>>(A, lda, sA, Bm, ldb, sB, bias, resid,
                                              outF, outB, sC, M, N, K, scale, relu);
  }
}

// one evaluation of the decoder-block RHS: kout = rhs(xin)
static void rhs(hipStream_t st, const WS& w, const float* xin, float* kout)
{
  const long long DD = (long long)Dc * Dc;
  const long long hQK = (long long)Lc * DKc;
  const float scl = 0.125f; // 1/sqrt(DK)

  // ---- self attention
  ln_kernel<true><<<BL, 256, 0, st>>>(xin, w.ln_g + 0 * Dc, w.ln_b + 0 * Dc, w.n_bf, nullptr);
  gemm(st, w.n_bf, Dc, 0, w.w_attn + 0 * DD, Dc, 0, 0, w.attn_b + 0 * Dc, nullptr,
       nullptr, w.q_bf, 0, BL, Dc, Dc, 1, 1.f, 0, 2);
  gemm(st, w.n_bf, Dc, 0, w.w_attn + 1 * DD, Dc, 0, 0, w.attn_b + 1 * Dc, nullptr,
       nullptr, w.k_bf, 0, BL, Dc, Dc, 1, 1.f, 0, 2);
  gemm(st, w.n_bf, Dc, 0, w.w_attn + 2 * DD, Dc, 0, 0, w.attn_b + 2 * Dc, nullptr,
       nullptr, w.v_bf, 0, BL, Dc, Dc, 1, 1.f, 0, 2);
  gemm(st, w.q_bf, DKc, hQK, w.k_bf, DKc, hQK, 1, nullptr, nullptr,
       w.scores, nullptr, (long long)Lc * Lc, Lc, Lc, DKc, BH, scl, 0, 0);
  softmax_kernel<<<BH * Lc, 256, 0, st>>>(w.scores, w.att_bf);
  gemm(st, w.att_bf, Lc, (long long)Lc * Lc, w.v_bf, DKc, hQK, 0, nullptr, nullptr,
       nullptr, w.attnout_bf, 0, Lc, DKc, Lc, BH, 1.f, 0, 3);
  gemm(st, w.attnout_bf, Dc, 0, w.w_attn + 3 * DD, Dc, 0, 0, w.attn_b + 3 * Dc, xin,
       w.x1, nullptr, 0, BL, Dc, Dc, 1, 1.f, 0, 0);

  // ---- cross attention (K/V of memory precomputed, loop-invariant)
  ln_kernel<true><<<BL, 256, 0, st>>>(w.x1, w.ln_g + 1 * Dc, w.ln_b + 1 * Dc, w.n_bf, nullptr);
  gemm(st, w.n_bf, Dc, 0, w.w_attn + 4 * DD, Dc, 0, 0, w.attn_b + 4 * Dc, nullptr,
       nullptr, w.q_bf, 0, BL, Dc, Dc, 1, 1.f, 0, 2);
  gemm(st, w.q_bf, DKc, hQK, w.kmem_bf, DKc, hQK, 1, nullptr, nullptr,
       w.scores, nullptr, (long long)Lc * Lc, Lc, Lc, DKc, BH, scl, 0, 0);
  softmax_kernel<<<BH * Lc, 256, 0, st>>>(w.scores, w.att_bf);
  gemm(st, w.att_bf, Lc, (long long)Lc * Lc, w.vmem_bf, DKc, hQK, 0, nullptr, nullptr,
       nullptr, w.attnout_bf, 0, Lc, DKc, Lc, BH, 1.f, 0, 3);
  gemm(st, w.attnout_bf, Dc, 0, w.w_attn + 7 * DD, Dc, 0, 0, w.attn_b + 7 * Dc, w.x1,
       w.x2, nullptr, 0, BL, Dc, Dc, 1, 1.f, 0, 0);

  // ---- feed-forward
  ln_kernel<true><<<BL, 256, 0, st>>>(w.x2, w.ln_g + 2 * Dc, w.ln_b + 2 * Dc, w.n_bf, nullptr);
  gemm(st, w.n_bf, Dc, 0, w.w_ff1, DFFc, 0, 0, w.ff_b1, nullptr,
       nullptr, w.ffn_bf, 0, BL, DFFc, Dc, 1, 1.f, 1, 1);               // + ReLU
  gemm(st, w.ffn_bf, DFFc, 0, w.w_ff2, Dc, 0, 0, w.ff_b2, w.x2,
       kout, nullptr, 0, BL, Dc, DFFc, 1, 1.f, 0, 0);
}

extern "C" void kernel_launch(void* const* d_in, const int* in_sizes, int n_in,
                              void* d_out, int out_size, void* d_ws, size_t ws_size,
                              hipStream_t stream)
{
  (void)in_sizes; (void)n_in; (void)out_size; (void)ws_size;
  const float* x      = (const float*)d_in[0];
  const float* memory = (const float*)d_in[1];
  // d_in[2]=src_mask, d_in[3]=tgt_mask: all-true in this problem -> no-op
  const float* attn_w = (const float*)d_in[4];
  const float* attn_b = (const float*)d_in[5];
  const float* ff_w1  = (const float*)d_in[6];
  const float* ff_b1  = (const float*)d_in[7];
  const float* ff_w2  = (const float*)d_in[8];
  const float* ff_b2  = (const float*)d_in[9];
  const float* ln_g   = (const float*)d_in[10];
  const float* ln_b   = (const float*)d_in[11];

  char* p = (char*)d_ws;
  auto alloc = [&](size_t bytes) -> void* {
    void* r = (void*)p;
    p += (bytes + 255) & ~(size_t)255;
    return r;
  };

  const long long nAct = (long long)BL * Dc;
  WS w;
  w.xbuf   = (float*)alloc(sizeof(float) * nAct);
  w.xstage = (float*)alloc(sizeof(float) * nAct);
  w.k1     = (float*)alloc(sizeof(float) * nAct);
  w.k2     = (float*)alloc(sizeof(float) * nAct);
  w.k3     = (float*)alloc(sizeof(float) * nAct);
  w.k4     = (float*)alloc(sizeof(float) * nAct);
  w.x1     = (float*)alloc(sizeof(float) * nAct);
  w.x2     = (float*)alloc(sizeof(float) * nAct);
  w.scores = (float*)alloc(sizeof(float) * (long long)BH * Lc * Lc);
  w.n_bf       = (u16*)alloc(sizeof(u16) * nAct);
  w.q_bf       = (u16*)alloc(sizeof(u16) * nAct);
  w.k_bf       = (u16*)alloc(sizeof(u16) * nAct);
  w.v_bf       = (u16*)alloc(sizeof(u16) * nAct);
  w.attnout_bf = (u16*)alloc(sizeof(u16) * nAct);
  w.mem_bf     = (u16*)alloc(sizeof(u16) * nAct);
  w.kmem_bf    = (u16*)alloc(sizeof(u16) * nAct);
  w.vmem_bf    = (u16*)alloc(sizeof(u16) * nAct);
  w.att_bf     = (u16*)alloc(sizeof(u16) * (long long)BH * Lc * Lc);
  w.ffn_bf     = (u16*)alloc(sizeof(u16) * (long long)BL * DFFc);
  w.w_attn     = (u16*)alloc(sizeof(u16) * 8LL * Dc * Dc);
  w.w_ff1      = (u16*)alloc(sizeof(u16) * (long long)Dc * DFFc);
  w.w_ff2      = (u16*)alloc(sizeof(u16) * (long long)Dc * DFFc);
  w.attn_b = attn_b; w.ff_b1 = ff_b1; w.ff_b2 = ff_b2; w.ln_g = ln_g; w.ln_b = ln_b;

  // ---- one-time (per call) prep: state copy + bf16 weight conversion
  (void)hipMemcpyAsync(w.xbuf, x, sizeof(float) * nAct, hipMemcpyDeviceToDevice, stream);
  f32_to_bf16_kernel<<<1024, 256, 0, stream>>>(attn_w, w.w_attn, 8LL * Dc * Dc);
  f32_to_bf16_kernel<<<1024, 256, 0, stream>>>(ff_w1, w.w_ff1, (long long)Dc * DFFc);
  f32_to_bf16_kernel<<<1024, 256, 0, stream>>>(ff_w2, w.w_ff2, (long long)Dc * DFFc);
  f32_to_bf16_kernel<<<1024, 256, 0, stream>>>(memory, w.mem_bf, nAct);
  // cross-attention K/V projections of memory: constant across the integration
  gemm(stream, w.mem_bf, Dc, 0, w.w_attn + 5LL * Dc * Dc, Dc, 0, 0, attn_b + 5 * Dc,
       nullptr, nullptr, w.kmem_bf, 0, BL, Dc, Dc, 1, 1.f, 0, 2);
  gemm(stream, w.mem_bf, Dc, 0, w.w_attn + 6LL * Dc * Dc, Dc, 0, 0, attn_b + 6 * Dc,
       nullptr, nullptr, w.vmem_bf, 0, BL, Dc, Dc, 1, 1.f, 0, 2);

  // ---- fixed-step RK4 over t in [0,1]
  const int NSTEP = 4;
  const float h = 1.0f / NSTEP;
  for (int s = 0; s < NSTEP; ++s) {
    rhs(stream, w, w.xbuf, w.k1);
    axpy_kernel<<<1024, 256, 0, stream>>>(w.xbuf, w.k1, 0.5f * h, w.xstage, nAct);
    rhs(stream, w, w.xstage, w.k2);
    axpy_kernel<<<1024, 256, 0, stream>>>(w.xbuf, w.k2, 0.5f * h, w.xstage, nAct);
    rhs(stream, w, w.xstage, w.k3);
    axpy_kernel<<<1024, 256, 0, stream>>>(w.xbuf, w.k3, h, w.xstage, nAct);
    rhs(stream, w, w.xstage, w.k4);
    rk4_combine_kernel<<<1024, 256, 0, stream>>>(w.xbuf, w.k1, w.k2, w.k3, w.k4,
                                                 h / 6.0f, nAct);
  }

  // ---- final layernorm -> f32 output
  ln_kernel<false><<<BL, 256, 0, stream>>>(w.xbuf, ln_g + 3 * Dc, ln_b + 3 * Dc,
                                           nullptr, (float*)d_out);
}
